// SchNet_1821066133918
// MI455X (gfx1250) — matrix operand order repaired and checked by reading, not compiled, not run on hardware
//
#include <hip/hip_runtime.h>
#include <hip/hip_bf16.h>

typedef __attribute__((ext_vector_type(16))) _Float16 v16h;
typedef __attribute__((ext_vector_type(8)))  _Float16 v8h;
typedef __attribute__((ext_vector_type(8)))  float    v8f;

#define DEV __device__ __forceinline__

static constexpr int HH = 128;   // hidden
static constexpr int FF = 128;   // filters
static constexpr int GG = 50;    // gaussians
static constexpr int GP = 64;    // gaussians padded to K multiple of 32
static constexpr int LL = 6;     // layers

static constexpr float CUTOFF = 5.0f;
static constexpr float LOG2C  = 0.6931471805599453f;
static constexpr float PIC    = 3.14159265358979323846f;

// ---------------------------------------------------------------- helpers

DEV float ssp(float x) {
  float ax = fabsf(x);
  return fmaxf(x, 0.0f) + log1pf(__expf(-ax)) - LOG2C;
}

DEV v8f wmma16(v16h a, v16h b, v8f c) {
  // D = A(16x32 f16) * B(32x16 f16) + C(16x16 f32)
  return __builtin_amdgcn_wmma_f32_16x16x32_f16(false, a, false, b, (short)0, c,
                                                false, false);
}

// A fragment (16x32, f16) from row-major f16 src, leading dim ld, k offset k0.
// ISA layout: lane m=lane&15, h=lane>>4; halves 0..7 = K k0+h*8 .. +7,
// halves 8..15 = K k0+16+h*8 .. +7  (two contiguous 16B reads).
DEV v16h a_frag_f16(const _Float16* src, int ld, int k0, int lane) {
  int m = lane & 15, h = lane >> 4;
  const _Float16* p = src + (size_t)m * ld + k0 + h * 8;
  v8h lo = *(const v8h*)(p);
  v8h hi = *(const v8h*)(p + 16);
  v16h a;
#pragma unroll
  for (int i = 0; i < 8; ++i) { a[i] = lo[i]; a[8 + i] = hi[i]; }
  return a;
}

// A fragment from row-major f32 src (full tile; converted to f16).
DEV v16h a_frag_f32(const float* src, int ld, int k0, int lane) {
  int m = lane & 15, h = lane >> 4;
  const float* p = src + (size_t)m * ld + k0 + h * 8;
  v16h a;
#pragma unroll
  for (int i = 0; i < 8; ++i) {
    a[i]     = (_Float16)p[i];
    a[8 + i] = (_Float16)p[16 + i];
  }
  return a;
}

// Straight vectorized copy of pre-swizzled weight fragments into LDS.
DEV void stage_copy(const v16h* __restrict__ g, v16h* __restrict__ lds,
                    int total, int tid, int nthr) {
  for (int idx = tid; idx < total; idx += nthr) lds[idx] = g[idx];
}

// ---------------------------------------------------------------- utility kernels

__global__ void k_copy(const float* __restrict__ s, float* __restrict__ d, int n) {
  int i = blockIdx.x * blockDim.x + threadIdx.x;
  if (i < n) d[i] = s[i];
}
__global__ void k_zero(float* __restrict__ p, int n) {
  int i = blockIdx.x * blockDim.x + threadIdx.x;
  if (i < n) p[i] = 0.0f;
}

// Convert fp32 [L, Ksrc, 128] weights into f16 B fragments in global memory,
// swizzled so fragment (ks, nt, lane) is one contiguous 32B v16h:
//   dst[l*per + (ks*8+nt)*32 + lane], with K padded to nksteps*32 with zeros.
__global__ void k_cvt_sw(const float* __restrict__ src, v16h* __restrict__ dst,
                         int nksteps, int kvalid, int srcLayerStride, int total) {
  int i = blockIdx.x * blockDim.x + threadIdx.x;
  if (i >= total) return;
  int per = nksteps * 8 * 32;
  int l = i / per, idx = i - l * per;
  int lane = idx & 31, nt = (idx >> 5) & 7, ks = idx >> 8;
  int n = lane & 15, kb = (lane >> 4) << 3;
  const float* s = src + (size_t)l * srcLayerStride;
  v16h b;
#pragma unroll
  for (int j = 0; j < 8; ++j) {
    int k0 = ks * 32 + kb + 2 * j + (j >= 4 ? 8 : 0), k1 = k0 + 1;
    b[2 * j]     = (k0 < kvalid) ? (_Float16)s[(size_t)k0 * 128 + nt * 16 + n]
                                 : (_Float16)0.0f;
    b[2 * j + 1] = (k1 < kvalid) ? (_Float16)s[(size_t)k1 * 128 + nt * 16 + n]
                                 : (_Float16)0.0f;
  }
  dst[i] = b;
}

// dist, cosine cutoff, padded RBF expansion (f16, vectorized stores) per edge
__global__ void k_prep(const float* __restrict__ pos, const float* __restrict__ off,
                       const int* __restrict__ edges, _Float16* __restrict__ demb,
                       float* __restrict__ Cenv, int E) {
  int e = blockIdx.x * blockDim.x + threadIdx.x;
  if (e >= E) return;
  int r = edges[e], c = edges[E + e];
  float dx = pos[c * 3 + 0] + off[(size_t)e * 3 + 0] - pos[r * 3 + 0];
  float dy = pos[c * 3 + 1] + off[(size_t)e * 3 + 1] - pos[r * 3 + 1];
  float dz = pos[c * 3 + 2] + off[(size_t)e * 3 + 2] - pos[r * 3 + 2];
  float dist = sqrtf(dx * dx + dy * dy + dz * dz);
  Cenv[e] = 0.5f * (__cosf(dist * (PIC / CUTOFF)) + 1.0f);
  const float step  = CUTOFF / (GG - 1);
  const float coeff = -0.5f / (step * step);
#pragma unroll
  for (int g8 = 0; g8 < GP; g8 += 8) {
    v8h pk;
#pragma unroll
    for (int j = 0; j < 8; ++j) {
      int g = g8 + j;
      float t = dist - (float)g * step;
      pk[j] = (g < GG) ? (_Float16)__expf(coeff * t * t) : (_Float16)0.0f;
    }
    *(v8h*)(demb + (size_t)e * GP + g8) = pk;
  }
}

// ---------------------------------------------------------------- hv = f16(v @ lin_w[l])
// Full 16-row tiles only; remainder rows handled by k_node_project_tail.

__global__ void __launch_bounds__(256)
k_node_project(const float* __restrict__ v, const v16h* __restrict__ linsw,
               _Float16* __restrict__ hv, int fullTiles) {
  __shared__ v16h sw[4 * 8 * 32];                 // 32 KB
  int tid = threadIdx.x, lane = tid & 31, wv = tid >> 5;
  stage_copy(linsw, sw, 4 * 8 * 32, tid, 256);
  __syncthreads();
  int tile = blockIdx.x * 8 + wv;
  if (tile >= fullTiles) return;                  // wave-uniform
  int base = tile * 16;
  v16h afr[4];
#pragma unroll
  for (int kk = 0; kk < 4; ++kk)
    afr[kk] = a_frag_f32(v + (size_t)base * HH, HH, kk * 32, lane);
  int n = lane & 15, m0 = (lane >> 4) * 8;
#pragma unroll
  for (int nt = 0; nt < 8; ++nt) {
    v8f acc;
#pragma unroll
    for (int i = 0; i < 8; ++i) acc[i] = 0.0f;
#pragma unroll
    for (int kk = 0; kk < 4; ++kk)
      acc = wmma16(afr[kk], sw[(kk * 8 + nt) * 32 + lane], acc);
#pragma unroll
    for (int r = 0; r < 8; ++r)
      hv[(size_t)(base + m0 + r) * FF + nt * 16 + n] = (_Float16)acc[r];
  }
}

__global__ void k_node_project_tail(const float* __restrict__ v,
                                    const float* __restrict__ lin,
                                    _Float16* __restrict__ hv, int start, int N) {
  int i = blockIdx.x * blockDim.x + threadIdx.x;
  int rows = N - start;
  if (i >= rows * FF) return;
  int row = start + i / FF, f = i % FF;
  float acc = 0.0f;
  for (int k = 0; k < HH; ++k)
    acc += v[(size_t)row * HH + k] * lin[(size_t)k * FF + f];
  hv[(size_t)row * FF + f] = (_Float16)acc;
}

// ------------------------------------------- fused filter-net + gather + scatter
// Full 16-edge tiles only; remainder edges handled by k_edge_tail.

__global__ void __launch_bounds__(256)
k_edge(const _Float16* __restrict__ demb, const float* __restrict__ Cenv,
       const int* __restrict__ edges, const _Float16* __restrict__ hv,
       const v16h* __restrict__ w1sw, const float* __restrict__ b1,
       const v16h* __restrict__ w2sw, const float* __restrict__ b2,
       float* __restrict__ agg, int E, int fullTiles) {
  __shared__ v16h sw1[2 * 8 * 32];                //  16 KB
  __shared__ v16h sw2[4 * 8 * 32];                //  32 KB
  __shared__ __align__(16) _Float16 scrA[8 * 16 * 32];  // 8 KB per-wave scratch
  int tid = threadIdx.x, lane = tid & 31, wv = tid >> 5;
  stage_copy(w1sw, sw1, 2 * 8 * 32, tid, 256);
  stage_copy(w2sw, sw2, 4 * 8 * 32, tid, 256);
  __syncthreads();
  int tile = blockIdx.x * 8 + wv;
  if (tile >= fullTiles) return;
  int ebase = tile * 16;
  int n = lane & 15, m0 = (lane >> 4) * 8;

  float Ce[8]; int re[8], ce[8];
#pragma unroll
  for (int r = 0; r < 8; ++r) {
    int e = ebase + m0 + r;
    Ce[r] = Cenv[e]; re[r] = edges[e]; ce[r] = edges[E + e];
  }
  // warm the random hv rows (256B each) while the filter-net GEMMs run
#pragma unroll
  for (int r = 0; r < 8; ++r) {
    __builtin_prefetch(hv + (size_t)re[r] * FF, 0, 3);
    __builtin_prefetch(hv + (size_t)re[r] * FF + 64, 0, 3);
  }

  // RBF A fragments for this 16-edge tile (K = 64 -> 2 k-steps)
  v16h ad0 = a_frag_f16(demb + (size_t)ebase * GP, GP, 0, lane);
  v16h ad1 = a_frag_f16(demb + (size_t)ebase * GP, GP, 32, lane);

  v8f acc2[8];
#pragma unroll
  for (int nt = 0; nt < 8; ++nt) {
    float bb = b2[nt * 16 + n];
#pragma unroll
    for (int i = 0; i < 8; ++i) acc2[nt][i] = bb;
  }

  _Float16* scr = scrA + wv * 16 * 32;
#pragma unroll
  for (int ks = 0; ks < 4; ++ks) {
#pragma unroll
    for (int t = 0; t < 2; ++t) {
      int nt1 = ks * 2 + t;
      v8f c;
      float bb = b1[nt1 * 16 + n];
#pragma unroll
      for (int i = 0; i < 8; ++i) c[i] = bb;
      c = wmma16(ad0, sw1[(0 * 8 + nt1) * 32 + lane], c);
      c = wmma16(ad1, sw1[(1 * 8 + nt1) * 32 + lane], c);
#pragma unroll
      for (int r = 0; r < 8; ++r)
        scr[(m0 + r) * 32 + t * 16 + n] = (_Float16)ssp(c[r]);
    }
    // same-wave LDS RAW is in order (DS in-order per ISA); re-read in A layout
    v16h at = a_frag_f16(scr, 32, 0, lane);
#pragma unroll
    for (int nt = 0; nt < 8; ++nt)
      acc2[nt] = wmma16(at, sw2[(ks * 8 + nt) * 32 + lane], acc2[nt]);
  }

  // e = hv[row] * Wf * C ; scatter-add into agg[col]
#pragma unroll
  for (int nt = 0; nt < 8; ++nt) {
    int coln = nt * 16 + n;
#pragma unroll
    for (int r = 0; r < 8; ++r) {
      float val = acc2[nt][r] * Ce[r] * (float)hv[(size_t)re[r] * FF + coln];
      atomicAdd(&agg[(size_t)ce[r] * FF + coln], val);
    }
  }
}

__global__ void k_edge_tail(const _Float16* __restrict__ demb,
                            const float* __restrict__ Cenv,
                            const int* __restrict__ edges,
                            const _Float16* __restrict__ hv,
                            const float* __restrict__ w1, const float* __restrict__ b1,
                            const float* __restrict__ w2, const float* __restrict__ b2,
                            float* __restrict__ agg, int start, int E) {
  int i = blockIdx.x * blockDim.x + threadIdx.x;
  int rem = E - start;
  if (i >= rem * FF) return;
  int e = start + i / FF, f = i % FF;
  float wf = b2[f];
  for (int j = 0; j < FF; ++j) {
    float t = b1[j];
    for (int g = 0; g < GG; ++g)
      t += (float)demb[(size_t)e * GP + g] * w1[(size_t)g * FF + j];
    wf += ssp(t) * w2[(size_t)j * FF + f];
  }
  int r = edges[e], c = edges[E + e];
  float val = (float)hv[(size_t)r * FF + f] * wf * Cenv[e];
  atomicAdd(&agg[(size_t)c * FF + f], val);
}

// ------------------------------------------- v += ssp(agg@u1+ub1)@u2 + ub2
// Full 16-row tiles only; remainder rows handled by k_node_update_tail.

__global__ void __launch_bounds__(256)
k_node_update(const float* __restrict__ agg, const v16h* __restrict__ u1sw,
              const float* __restrict__ ub1, const v16h* __restrict__ u2sw,
              const float* __restrict__ ub2, float* __restrict__ v, int fullTiles) {
  __shared__ v16h su2[4 * 8 * 32];                // 32 KB
  __shared__ __align__(16) _Float16 scrA[8 * 16 * 32];  // 8 KB
  int tid = threadIdx.x, lane = tid & 31, wv = tid >> 5;
  stage_copy(u2sw, su2, 4 * 8 * 32, tid, 256);
  __syncthreads();
  int tile = blockIdx.x * 8 + wv;
  if (tile >= fullTiles) return;
  int base = tile * 16;
  int n = lane & 15, m0 = (lane >> 4) * 8;

  v16h afr[4];
#pragma unroll
  for (int kk = 0; kk < 4; ++kk)
    afr[kk] = a_frag_f32(agg + (size_t)base * FF, FF, kk * 32, lane);

  v8f acc2[8];
#pragma unroll
  for (int nt = 0; nt < 8; ++nt) {
    float bb = ub2[nt * 16 + n];
#pragma unroll
    for (int i = 0; i < 8; ++i) acc2[nt][i] = bb;
  }

  _Float16* scr = scrA + wv * 16 * 32;
#pragma unroll
  for (int ks = 0; ks < 4; ++ks) {
#pragma unroll
    for (int t = 0; t < 2; ++t) {
      int nt1 = ks * 2 + t;
      v8f c;
      float bb = ub1[nt1 * 16 + n];
#pragma unroll
      for (int i = 0; i < 8; ++i) c[i] = bb;
#pragma unroll
      for (int kk = 0; kk < 4; ++kk)   // u1 fragments direct from swizzled global
        c = wmma16(afr[kk], u1sw[(kk * 8 + nt1) * 32 + lane], c);
#pragma unroll
      for (int r = 0; r < 8; ++r)
        scr[(m0 + r) * 32 + t * 16 + n] = (_Float16)ssp(c[r]);
    }
    v16h at = a_frag_f16(scr, 32, 0, lane);
#pragma unroll
    for (int nt = 0; nt < 8; ++nt)
      acc2[nt] = wmma16(at, su2[(ks * 8 + nt) * 32 + lane], acc2[nt]);
  }

#pragma unroll
  for (int nt = 0; nt < 8; ++nt) {
    int coln = nt * 16 + n;
#pragma unroll
    for (int r = 0; r < 8; ++r)
      v[(size_t)(base + m0 + r) * HH + coln] += acc2[nt][r];
  }
}

__global__ void k_node_update_tail(const float* __restrict__ agg,
                                   const float* __restrict__ u1,
                                   const float* __restrict__ ub1,
                                   const float* __restrict__ u2,
                                   const float* __restrict__ ub2,
                                   float* __restrict__ v, int start, int N) {
  int i = blockIdx.x * blockDim.x + threadIdx.x;
  int rows = N - start;
  if (i >= rows * HH) return;
  int row = start + i / HH, f = i % HH;
  float out = ub2[f];
  for (int j = 0; j < HH; ++j) {
    float t = ub1[j];
    for (int k = 0; k < FF; ++k)
      t += agg[(size_t)row * FF + k] * u1[(size_t)k * HH + j];
    out += ssp(t) * u2[(size_t)j * HH + f];
  }
  v[(size_t)row * HH + f] += out;
}

// ---------------------------------------------------------------- launcher

extern "C" void kernel_launch(void* const* d_in, const int* in_sizes, int n_in,
                              void* d_out, int out_size, void* d_ws, size_t ws_size,
                              hipStream_t stream) {
  (void)n_in; (void)out_size; (void)ws_size;
  const float* v0    = (const float*)d_in[0];
  const float* pos   = (const float*)d_in[1];
  const float* offr  = (const float*)d_in[2];
  const int*   edges = (const int*)  d_in[3];
  const float* lin_w = (const float*)d_in[4];
  const float* w1    = (const float*)d_in[5];
  const float* b1    = (const float*)d_in[6];
  const float* w2    = (const float*)d_in[7];
  const float* b2    = (const float*)d_in[8];
  const float* u1    = (const float*)d_in[9];
  const float* ub1   = (const float*)d_in[10];
  const float* u2    = (const float*)d_in[11];
  const float* ub2   = (const float*)d_in[12];

  const int N = in_sizes[0] / HH;
  const int E = in_sizes[3] / 2;
  float* vout = (float*)d_out;

  // workspace carve-out (256B aligned)
  char* ws = (char*)d_ws;
  auto carve = [&](size_t bytes) {
    char* p = ws;
    ws += (bytes + 255) & ~(size_t)255;
    return p;
  };
  const int PER2 = 2 * 8 * 32;   // fragments per layer, K=64
  const int PER4 = 4 * 8 * 32;   // fragments per layer, K=128
  _Float16* demb16 = (_Float16*)carve((size_t)E * GP * 2);
  float*    Cenv   = (float*)   carve((size_t)E * 4);
  _Float16* hv16   = (_Float16*)carve((size_t)N * FF * 2);
  float*    agg    = (float*)   carve((size_t)N * FF * 4);
  v16h*     w1sw   = (v16h*)    carve((size_t)LL * PER2 * sizeof(v16h));
  v16h*     w2sw   = (v16h*)    carve((size_t)LL * PER4 * sizeof(v16h));
  v16h*     linsw  = (v16h*)    carve((size_t)LL * PER4 * sizeof(v16h));
  v16h*     u1sw   = (v16h*)    carve((size_t)LL * PER4 * sizeof(v16h));
  v16h*     u2sw   = (v16h*)    carve((size_t)LL * PER4 * sizeof(v16h));

  const int T = 256;
  k_copy<<<(N * HH + T - 1) / T, T, 0, stream>>>(v0, vout, N * HH);
  k_prep<<<(E + T - 1) / T, T, 0, stream>>>(pos, offr, edges, demb16, Cenv, E);

  const int tot2 = LL * PER2, tot4 = LL * PER4;
  k_cvt_sw<<<(tot2 + T - 1) / T, T, 0, stream>>>(w1,    w1sw,  2, GG,  GG * FF, tot2);
  k_cvt_sw<<<(tot4 + T - 1) / T, T, 0, stream>>>(w2,    w2sw,  4, FF,  FF * FF, tot4);
  k_cvt_sw<<<(tot4 + T - 1) / T, T, 0, stream>>>(lin_w, linsw, 4, HH,  HH * FF, tot4);
  k_cvt_sw<<<(tot4 + T - 1) / T, T, 0, stream>>>(u1,    u1sw,  4, FF,  FF * HH, tot4);
  k_cvt_sw<<<(tot4 + T - 1) / T, T, 0, stream>>>(u2,    u2sw,  4, HH,  HH * HH, tot4);

  const int fullNT = N >> 4, remN = N & 15, startN = fullNT * 16;
  const int fullET = E >> 4, remE = E & 15, startE = fullET * 16;
  const int nodeBlocks = (fullNT + 7) / 8;
  const int edgeBlocks = (fullET + 7) / 8;

  for (int l = 0; l < LL; ++l) {
    if (fullNT)
      k_node_project<<<nodeBlocks, T, 0, stream>>>(vout, linsw + (size_t)l * PER4,
                                                   hv16, fullNT);
    if (remN)
      k_node_project_tail<<<(remN * FF + T - 1) / T, T, 0, stream>>>(
          vout, lin_w + (size_t)l * HH * FF, hv16, startN, N);

    k_zero<<<(N * FF + T - 1) / T, T, 0, stream>>>(agg, N * FF);

    if (fullET)
      k_edge<<<edgeBlocks, T, 0, stream>>>(
          demb16, Cenv, edges, hv16,
          w1sw + (size_t)l * PER2, b1 + (size_t)l * FF,
          w2sw + (size_t)l * PER4, b2 + (size_t)l * FF, agg, E, fullET);
    if (remE)
      k_edge_tail<<<(remE * FF + T - 1) / T, T, 0, stream>>>(
          demb16, Cenv, edges, hv16,
          w1 + (size_t)l * GG * FF, b1 + (size_t)l * FF,
          w2 + (size_t)l * FF * FF, b2 + (size_t)l * FF, agg, startE, E);

    if (fullNT)
      k_node_update<<<nodeBlocks, T, 0, stream>>>(
          agg, u1sw + (size_t)l * PER4, ub1 + (size_t)l * HH,
          u2sw + (size_t)l * PER4, ub2 + (size_t)l * HH, vout, fullNT);
    if (remN)
      k_node_update_tail<<<(remN * HH + T - 1) / T, T, 0, stream>>>(
          agg, u1 + (size_t)l * FF * HH, ub1 + (size_t)l * HH,
          u2 + (size_t)l * HH * HH, ub2 + (size_t)l * HH, vout, startN, N);
  }
}
